// BLTModel_13872744366807
// MI455X (gfx1250) — matrix operand order repaired and checked by reading, not compiled, not run on hardware
//
#include <hip/hip_runtime.h>
#include <hip/hip_bf16.h>
#include <math.h>

// ---------------- types / constants ----------------
typedef __attribute__((ext_vector_type(16))) _Float16 v16h;
typedef __attribute__((ext_vector_type(8)))  _Float16 v8h;
typedef __attribute__((ext_vector_type(4)))  _Float16 v4h;
typedef __attribute__((ext_vector_type(8)))  float    v8f;

#define B_   2
#define S_   4096
#define P_   256
#define H_   1024
#define L_   4
#define V_   256
#define NH_  16
#define CAH_ 8

union AF { v16h v; v8h h[2]; };

// ---------------- fp32 -> fp16 convert (vector) ----------------
__global__ void k_cvt(const float* __restrict__ in, _Float16* __restrict__ out, int n4) {
    int i = blockIdx.x * 256 + threadIdx.x;
    if (i < n4) {
        float4 v = ((const float4*)in)[i];
        v4h r = { (_Float16)v.x, (_Float16)v.y, (_Float16)v.z, (_Float16)v.w };
        ((v4h*)out)[i] = r;
    }
}

// ---------------- embedding gather ----------------
__global__ __launch_bounds__(256) void k_embed(const int* __restrict__ bytes,
                                               const float* __restrict__ emb,
                                               float* __restrict__ x) {
    int row = blockIdx.x;                        // b*S + s
    int by  = bytes[row];
    float4 e = ((const float4*)(emb + (size_t)by * H_))[threadIdx.x];
    ((float4*)(x + (size_t)row * H_))[threadIdx.x] = e;
}

// ---------------- ragged patch mean-pool (contiguous ranges, no atomics) ----
__global__ __launch_bounds__(256) void k_pool(const int* __restrict__ bytes,
                                              const int* __restrict__ bd,
                                              const float* __restrict__ emb,
                                              float* __restrict__ patches) {
    int p = blockIdx.x, b = blockIdx.y;
    const int* bdb = bd + b * P_;
    int lo = (p == 0)       ? 0  : bdb[p - 1];
    int hi = (p == P_ - 1)  ? S_ : bdb[p];
    if (lo < 0) lo = 0;
    if (hi > S_) hi = S_;
    float ax = 0.f, ay = 0.f, az = 0.f, aw = 0.f;
    for (int i = lo; i < hi; ++i) {
        int by = bytes[b * S_ + i];
        float4 e = ((const float4*)(emb + (size_t)by * H_))[threadIdx.x];
        ax += e.x; ay += e.y; az += e.z; aw += e.w;
    }
    float inv = 1.f / fmaxf((float)(hi - lo), 1.f);
    float4 r = { ax * inv, ay * inv, az * inv, aw * inv };
    ((float4*)(patches + (size_t)(b * P_ + p) * H_))[threadIdx.x] = r;
}

// ---------------- layernorm (row per block, H=1024) ----------------
__global__ __launch_bounds__(256) void k_ln(const float* __restrict__ in,
                                            const float* __restrict__ g,
                                            const float* __restrict__ bta,
                                            float* outf, _Float16* outh) {
    int row = blockIdx.x, tid = threadIdx.x;
    const float* xr = in + (size_t)row * H_;
    float4 v = ((const float4*)xr)[tid];
    float s  = v.x + v.y + v.z + v.w;
    float ss = v.x * v.x + v.y * v.y + v.z * v.z + v.w * v.w;
    #pragma unroll
    for (int o = 16; o; o >>= 1) { s += __shfl_xor(s, o, 32); ss += __shfl_xor(ss, o, 32); }
    __shared__ float red0[8], red1[8];
    if ((tid & 31) == 0) { red0[tid >> 5] = s; red1[tid >> 5] = ss; }
    __syncthreads();
    float st = 0.f, sst = 0.f;
    #pragma unroll
    for (int i = 0; i < 8; ++i) { st += red0[i]; sst += red1[i]; }
    float mu   = st * (1.f / (float)H_);
    float var  = sst * (1.f / (float)H_) - mu * mu;
    float rstd = rsqrtf(var + 1e-6f);
    float4 gv = ((const float4*)g)[tid];
    float4 bv = ((const float4*)bta)[tid];
    float y0 = (v.x - mu) * rstd * gv.x + bv.x;
    float y1 = (v.y - mu) * rstd * gv.y + bv.y;
    float y2 = (v.z - mu) * rstd * gv.z + bv.z;
    float y3 = (v.w - mu) * rstd * gv.w + bv.w;
    if (outf) { float4 r = { y0, y1, y2, y3 }; ((float4*)(outf + (size_t)row * H_))[tid] = r; }
    if (outh) { v4h r = { (_Float16)y0, (_Float16)y1, (_Float16)y2, (_Float16)y3 };
                ((v4h*)(outh + (size_t)row * H_))[tid] = r; }
}

// ---------------- WMMA f16 GEMM: out = act(A@W + bias) [+ resid] -----------
// A:[M,K] f16 (lda), W:[K,N] f16 slice (ldw), block tile 64x128, wave 16x64.
// Epilogue flags are compile-time; staging uses strength-reduced pointer
// increments and unconditional 2-tile-ahead speculative prefetch.
#define TM 64
#define TN 128
#define TK 32
template<int GELU, int RESID, int OUTF, int OUTH>
__global__ __launch_bounds__(256) void k_gemm(const _Float16* __restrict__ A, int lda,
                                              const _Float16* __restrict__ W, int ldw,
                                              const float* __restrict__ bias,
                                              const float* resid,
                                              float* outf, _Float16* outh,
                                              int N, int K) {
    __shared__ _Float16 As[TM][TK + 8];    // row-major A tile
    __shared__ _Float16 Bt[TN][TK + 8];    // transposed W tile: Bt[n][k]
    const int tid  = threadIdx.x;
    const int wave = tid >> 5, lane = tid & 31;
    const int wm = wave & 3, wn = wave >> 2;           // 4 x 2 wave grid
    const int bm = blockIdx.y * TM, bn = blockIdx.x * TN;
    v8f acc[4] = {};

    // --- staging addresses, hoisted; advance by constant strides per step ---
    const int ar  = tid >> 2;              // A tile row   (64 rows, 4 thr/row)
    const int ac  = (tid & 3) * 8;         // A tile col   (v8h)
    const _Float16* aptr = A + (size_t)(bm + ar) * lda + ac;
    _Float16* asdst = &As[ar][ac];
    const int kk0 = tid >> 4,        nn0 = (tid & 15) * 8;   // W rows 0..15
    const int kk1 = 16 + (tid >> 4), nn1 = nn0;              // W rows 16..31
    const _Float16* wptr0 = W + (size_t)kk0 * ldw + bn + nn0;
    const _Float16* wptr1 = W + (size_t)kk1 * ldw + bn + nn1;
    const size_t wstep = (size_t)TK * ldw;

    // --- per-wave fragment LDS addresses, hoisted ---
    const int arow = wm * 16 + (lane & 15);
    const int ha   = (lane >> 4) << 3;                 // 0 or 8
    const int kb   = (lane >> 4) << 4;                 // 0 or 16
    const _Float16* afr0 = &As[arow][ha];
    const _Float16* afr1 = &As[arow][16 + ha];

    for (int k0 = 0; k0 < K; k0 += TK) {
        *(v8h*)asdst = *(const v8h*)aptr;
        __builtin_prefetch(aptr + 2 * TK, 0, 1);       // global_prefetch_b8
        v8h w0 = *(const v8h*)wptr0;
        v8h w1 = *(const v8h*)wptr1;
        __builtin_prefetch(wptr0 + 2 * wstep, 0, 1);
        __builtin_prefetch(wptr1 + 2 * wstep, 0, 1);
        #pragma unroll
        for (int e = 0; e < 8; ++e) {                  // transpose into LDS
            Bt[nn0 + e][kk0] = w0[e];
            Bt[nn1 + e][kk1] = w1[e];
        }
        __syncthreads();
        AF af;
        af.h[0] = *(const v8h*)afr0;                   // K = ha..ha+7
        af.h[1] = *(const v8h*)afr1;                   // K = 16+ha..
        #pragma unroll
        for (int t = 0; t < 4; ++t) {
            const int brow = wn * 64 + t * 16 + (lane & 15);
            AF bf;
            bf.h[0] = *(const v8h*)&Bt[brow][ha];
            bf.h[1] = *(const v8h*)&Bt[brow][16 + ha];
            acc[t] = __builtin_amdgcn_wmma_f32_16x16x32_f16(
                false, af.v, false, bf.v, (short)0, acc[t], false, false);
        }
        __syncthreads();
        aptr  += TK;
        wptr0 += wstep;
        wptr1 += wstep;
    }
    // epilogue: lane (row half, col) per ISA C/D layout — straight-line code
    const int erow = (lane >> 4) << 3;                 // 0 or 8
    const int ecol = lane & 15;
    #pragma unroll
    for (int t = 0; t < 4; ++t) {
        int n = bn + wn * 64 + t * 16 + ecol;
        float bv = bias[n];
        #pragma unroll
        for (int r = 0; r < 8; ++r) {
            int m = bm + wm * 16 + erow + r;
            float y = acc[t][r] + bv;
            if (GELU)  y = 0.5f * y * (1.f + erff(y * 0.70710678118654752f));
            if (RESID) y += resid[(size_t)m * N + n];
            if (OUTF)  outf[(size_t)m * N + n] = y;
            if (OUTH)  outh[(size_t)m * N + n] = (_Float16)y;
        }
    }
}

// ---------------- global causal attention (16 heads, dh=64) ----------------
__global__ __launch_bounds__(256) void k_attn_g(const float* __restrict__ qkv,
                                                _Float16* __restrict__ out) {
    const int lane = threadIdx.x & 31;
    const int qrow = blockIdx.x * 8 + (threadIdx.x >> 5);
    const int h = blockIdx.y, b = blockIdx.z;
    const float* qp = qkv + (size_t)(b * P_ + qrow) * 3 * H_ + h * 64;
    float q0 = qp[lane], q1 = qp[lane + 32];
    float m = -1e30f, l = 0.f, a0 = 0.f, a1 = 0.f;
    for (int j = 0; j <= qrow; ++j) {
        const float* kp = qkv + (size_t)(b * P_ + j) * 3 * H_ + H_ + h * 64;
        float d = q0 * kp[lane] + q1 * kp[lane + 32];
        #pragma unroll
        for (int o = 16; o; o >>= 1) d += __shfl_xor(d, o, 32);
        d *= 0.125f;                                   // 1/sqrt(64)
        float mn = fmaxf(m, d);
        float c = expf(m - mn), w = expf(d - mn);
        const float* vp = qkv + (size_t)(b * P_ + j) * 3 * H_ + 2 * H_ + h * 64;
        l = l * c + w;
        a0 = a0 * c + w * vp[lane];
        a1 = a1 * c + w * vp[lane + 32];
        m = mn;
    }
    float inv = 1.f / l;
    _Float16* op = out + (size_t)(b * P_ + qrow) * H_ + h * 64;
    op[lane]      = (_Float16)(a0 * inv);
    op[lane + 32] = (_Float16)(a1 * inv);
}

// ---------------- cross attention (8 heads, dh=128, no mask) ---------------
__global__ __launch_bounds__(256) void k_attn_c(const float* __restrict__ q,
                                                const float* __restrict__ k,
                                                const float* __restrict__ v,
                                                _Float16* __restrict__ out) {
    const int lane = threadIdx.x & 31;
    const int qrow = blockIdx.x * 8 + (threadIdx.x >> 5);
    const int h = blockIdx.y, b = blockIdx.z;
    const float* qp = q + (size_t)(b * S_ + qrow) * H_ + h * 128;
    float q0 = qp[lane], q1 = qp[lane + 32], q2 = qp[lane + 64], q3 = qp[lane + 96];
    float m = -1e30f, l = 0.f, a0 = 0.f, a1 = 0.f, a2 = 0.f, a3 = 0.f;
    for (int j = 0; j < P_; ++j) {
        const float* kp = k + (size_t)(b * P_ + j) * H_ + h * 128;
        float d = q0 * kp[lane] + q1 * kp[lane + 32] + q2 * kp[lane + 64] + q3 * kp[lane + 96];
        #pragma unroll
        for (int o = 16; o; o >>= 1) d += __shfl_xor(d, o, 32);
        d *= 0.08838834764831845f;                     // 1/sqrt(128)
        float mn = fmaxf(m, d);
        float c = expf(m - mn), w = expf(d - mn);
        const float* vp = v + (size_t)(b * P_ + j) * H_ + h * 128;
        l = l * c + w;
        a0 = a0 * c + w * vp[lane];
        a1 = a1 * c + w * vp[lane + 32];
        a2 = a2 * c + w * vp[lane + 64];
        a3 = a3 * c + w * vp[lane + 96];
        m = mn;
    }
    float inv = 1.f / l;
    _Float16* op = out + (size_t)(b * S_ + qrow) * H_ + h * 128;
    op[lane]      = (_Float16)(a0 * inv);
    op[lane + 32] = (_Float16)(a1 * inv);
    op[lane + 64] = (_Float16)(a2 * inv);
    op[lane + 96] = (_Float16)(a3 * inv);
}

// ---------------- host orchestration ----------------
extern "C" void kernel_launch(void* const* d_in, const int* in_sizes, int n_in,
                              void* d_out, int out_size, void* d_ws, size_t ws_size,
                              hipStream_t stream) {
    (void)in_sizes; (void)n_in; (void)out_size; (void)ws_size;
    const int*   byte_seq = (const int*)d_in[0];
    const int*   bd       = (const int*)d_in[1];
    const float* emb      = (const float*)d_in[2];
    const float* g_ln1_g  = (const float*)d_in[3];
    const float* g_ln1_b  = (const float*)d_in[4];
    const float* g_wqkv   = (const float*)d_in[5];
    const float* g_bqkv   = (const float*)d_in[6];
    const float* g_wo     = (const float*)d_in[7];
    const float* g_bo     = (const float*)d_in[8];
    const float* g_ln2_g  = (const float*)d_in[9];
    const float* g_ln2_b  = (const float*)d_in[10];
    const float* g_w1     = (const float*)d_in[11];
    const float* g_b1     = (const float*)d_in[12];
    const float* g_w2     = (const float*)d_in[13];
    const float* g_b2     = (const float*)d_in[14];
    const float* fn_g     = (const float*)d_in[15];
    const float* fn_b     = (const float*)d_in[16];
    const float* ca_ln_g  = (const float*)d_in[17];
    const float* ca_ln_b  = (const float*)d_in[18];
    const float* ca_wqkv  = (const float*)d_in[19];
    const float* ca_bqkv  = (const float*)d_in[20];
    const float* ca_wo    = (const float*)d_in[21];
    const float* ca_bo    = (const float*)d_in[22];
    const float* head_w   = (const float*)d_in[23];
    const float* head_b   = (const float*)d_in[24];

    char* wsp = (char*)d_ws;
    size_t off = 0;
    auto alloc = [&](size_t bytes) -> char* {
        char* p = wsp + off;
        off = (off + bytes + 255) & ~(size_t)255;
        return p;
    };
    _Float16* wqkv_h   = (_Float16*)alloc((size_t)L_ * H_ * 3 * H_ * 2);
    _Float16* wo_h     = (_Float16*)alloc((size_t)L_ * H_ * H_ * 2);
    _Float16* w1_h     = (_Float16*)alloc((size_t)L_ * H_ * 4 * H_ * 2);
    _Float16* w2_h     = (_Float16*)alloc((size_t)L_ * 4 * H_ * H_ * 2);
    _Float16* cawqkv_h = (_Float16*)alloc((size_t)H_ * 3 * H_ * 2);
    _Float16* cawo_h   = (_Float16*)alloc((size_t)H_ * H_ * 2);
    _Float16* whead_h  = (_Float16*)alloc((size_t)H_ * V_ * 2);
    float*    x        = (float*)alloc((size_t)B_ * S_ * H_ * 4);
    float*    patches  = (float*)alloc((size_t)B_ * P_ * H_ * 4);
    _Float16* pn_h     = (_Float16*)alloc((size_t)B_ * P_ * H_ * 2);
    float*    qkv_g    = (float*)alloc((size_t)B_ * P_ * 3 * H_ * 4);
    _Float16* attng_h  = (_Float16*)alloc((size_t)B_ * P_ * H_ * 2);
    _Float16* ffn1_h   = (_Float16*)alloc((size_t)B_ * P_ * 4 * H_ * 2);
    float*    pfn      = (float*)alloc((size_t)B_ * P_ * H_ * 4);
    _Float16* kvn_h    = (_Float16*)alloc((size_t)B_ * P_ * H_ * 2);
    _Float16* qn_h     = (_Float16*)alloc((size_t)B_ * S_ * H_ * 2);
    float*    qb       = (float*)alloc((size_t)B_ * S_ * H_ * 4);
    float*    kb       = (float*)alloc((size_t)B_ * P_ * H_ * 4);
    float*    vb       = (float*)alloc((size_t)B_ * P_ * H_ * 4);
    _Float16* attnc_h  = (_Float16*)alloc((size_t)B_ * S_ * H_ * 2);
    _Float16* hf16     = (_Float16*)alloc((size_t)B_ * S_ * H_ * 2);

    auto cvt = [&](const float* src, _Float16* dst, size_t n) {
        int n4 = (int)(n / 4);
        k_cvt<<<(n4 + 255) / 256, 256, 0, stream>>>(src, dst, n4);
    };
    cvt(g_wqkv,  wqkv_h,   (size_t)L_ * H_ * 3 * H_);
    cvt(g_wo,    wo_h,     (size_t)L_ * H_ * H_);
    cvt(g_w1,    w1_h,     (size_t)L_ * H_ * 4 * H_);
    cvt(g_w2,    w2_h,     (size_t)L_ * 4 * H_ * H_);
    cvt(ca_wqkv, cawqkv_h, (size_t)H_ * 3 * H_);
    cvt(ca_wo,   cawo_h,   (size_t)H_ * H_);
    cvt(head_w,  whead_h,  (size_t)H_ * V_);

    // plain: out_f32 = A@W + b
    auto gemm_f = [&](const _Float16* A, int lda, const _Float16* W, int ldw,
                      const float* bias, float* outf, int M, int N, int K) {
        dim3 g(N / TN, M / TM);
        k_gemm<0,0,1,0><<<g, 256, 0, stream>>>(A, lda, W, ldw, bias, nullptr, outf, nullptr, N, K);
    };
    // residual: out_f32 = resid + A@W + b   (out may alias resid; one R/W per elem)
    auto gemm_rf = [&](const _Float16* A, int lda, const _Float16* W, int ldw,
                       const float* bias, const float* resid, float* outf,
                       int M, int N, int K) {
        dim3 g(N / TN, M / TM);
        k_gemm<0,1,1,0><<<g, 256, 0, stream>>>(A, lda, W, ldw, bias, resid, outf, nullptr, N, K);
    };
    // gelu -> f16: out_f16 = gelu(A@W + b)
    auto gemm_gh = [&](const _Float16* A, int lda, const _Float16* W, int ldw,
                       const float* bias, _Float16* outh, int M, int N, int K) {
        dim3 g(N / TN, M / TM);
        k_gemm<1,0,0,1><<<g, 256, 0, stream>>>(A, lda, W, ldw, bias, nullptr, nullptr, outh, N, K);
    };
    // residual -> f16: out_f16 = resid + A@W + b
    auto gemm_rh = [&](const _Float16* A, int lda, const _Float16* W, int ldw,
                       const float* bias, const float* resid, _Float16* outh,
                       int M, int N, int K) {
        dim3 g(N / TN, M / TM);
        k_gemm<0,1,0,1><<<g, 256, 0, stream>>>(A, lda, W, ldw, bias, resid, nullptr, outh, N, K);
    };

    k_embed<<<B_ * S_, 256, 0, stream>>>(byte_seq, emb, x);
    k_pool<<<dim3(P_, B_), 256, 0, stream>>>(byte_seq, bd, emb, patches);

    for (int l = 0; l < L_; ++l) {
        const _Float16* Wqkv = wqkv_h + (size_t)l * H_ * 3 * H_;
        const _Float16* Wo   = wo_h   + (size_t)l * H_ * H_;
        const _Float16* W1   = w1_h   + (size_t)l * H_ * 4 * H_;
        const _Float16* W2   = w2_h   + (size_t)l * 4 * H_ * H_;
        k_ln<<<B_ * P_, 256, 0, stream>>>(patches, g_ln1_g + l * H_, g_ln1_b + l * H_, nullptr, pn_h);
        gemm_f(pn_h, H_, Wqkv, 3 * H_, g_bqkv + l * 3 * H_, qkv_g, B_ * P_, 3 * H_, H_);
        k_attn_g<<<dim3(P_ / 8, NH_, B_), 256, 0, stream>>>(qkv_g, attng_h);
        gemm_rf(attng_h, H_, Wo, H_, g_bo + l * H_, patches, patches, B_ * P_, H_, H_);
        k_ln<<<B_ * P_, 256, 0, stream>>>(patches, g_ln2_g + l * H_, g_ln2_b + l * H_, nullptr, pn_h);
        gemm_gh(pn_h, H_, W1, 4 * H_, g_b1 + l * 4 * H_, ffn1_h, B_ * P_, 4 * H_, H_);
        gemm_rf(ffn1_h, 4 * H_, W2, H_, g_b2 + l * H_, patches, patches, B_ * P_, H_, 4 * H_);
    }

    // final norm on patches, then cross-attn LN on both streams
    k_ln<<<B_ * P_, 256, 0, stream>>>(patches, fn_g, fn_b, pfn, nullptr);
    k_ln<<<B_ * P_, 256, 0, stream>>>(pfn, ca_ln_g, ca_ln_b, nullptr, kvn_h);
    k_ln<<<B_ * S_, 256, 0, stream>>>(x, ca_ln_g, ca_ln_b, nullptr, qn_h);

    gemm_f(qn_h,  H_, cawqkv_h,          3 * H_, ca_bqkv,          qb, B_ * S_, H_, H_);
    gemm_f(kvn_h, H_, cawqkv_h + H_,     3 * H_, ca_bqkv + H_,     kb, B_ * P_, H_, H_);
    gemm_f(kvn_h, H_, cawqkv_h + 2 * H_, 3 * H_, ca_bqkv + 2 * H_, vb, B_ * P_, H_, H_);

    k_attn_c<<<dim3(S_ / 8, CAH_, B_), 256, 0, stream>>>(qb, kb, vb, attnc_h);

    // h = x + attn@ca_wo + ca_bo  (write f16 for head GEMM)
    gemm_rh(attnc_h, H_, cawo_h, H_, ca_bo, x, hf16, B_ * S_, H_, H_);
    // logits = h @ head_w + head_b
    gemm_f(hf16, H_, whead_h, V_, head_b, (float*)d_out, B_ * S_, V_, H_);
}
